// SMoREGate_20057497272798
// MI455X (gfx1250) — matrix operand order repaired and checked by reading, not compiled, not run on hardware
//
#include <hip/hip_runtime.h>
#include <hip/hip_bf16.h>
#include <stdint.h>

typedef __attribute__((ext_vector_type(16))) __bf16 v16bf;
typedef __attribute__((ext_vector_type(8)))  float  v8f;
typedef __attribute__((ext_vector_type(2)))  float  v2f;

#define T_TOK 16384
#define DIM   2048
#define NE    64
#define NA    8

// ---- workspace byte offsets (total ~290 MB) ----
#define WS_XB   0ull           // x bf16         16384*2048*2
#define WS_W1B  67108864ull    // W1 bf16        2048*2048*2
#define WS_W2B  75497472ull    // W2 bf16
#define WS_HB   83886080ull    // h  bf16        16384*2048*2
#define WS_P    150994944ull   // p  f32         16384*2048*4
#define WS_KN   285212672ull   // keys normalized f32  64*2048*4
#define WS_LG   285736960ull   // logits f32     16384*64*4
#define WS_RN   289931264ull   // row sumsq f32  16384*4

// ---- d_out element offsets (f32/int32 slots) ----
#define O_IDX 0
#define O_SC  131072
#define O_BL  262144
#define O_LD  262145
#define O_IMP 262209
#define O_SK  262273ull

// ------------------------------------------------------------------
// CDNA5 async memory->LDS DMA (ASYNCcnt-tracked), inline asm path
__device__ __forceinline__ void async_load_b128(unsigned lds_off,
                                                const void* gaddr) {
  asm volatile("global_load_async_to_lds_b128 %0, %1, off"
               :: "v"(lds_off), "v"(gaddr) : "memory");
}
#define WAIT_ASYNC_LE4() asm volatile("s_wait_asynccnt 0x4" ::: "memory")
#define WAIT_ASYNC_0()   asm volatile("s_wait_asynccnt 0x0" ::: "memory")

// ------------------------------------------------------------------
// f32 -> bf16 convert
__global__ void cvt_bf16_kernel(const float* __restrict__ src,
                                __bf16* __restrict__ dst, int n) {
  int i = blockIdx.x * blockDim.x + threadIdx.x;
  if (i < n) dst[i] = (__bf16)src[i];
}

// ------------------------------------------------------------------
// normalize keys: kn[e] = keys[e] / max(||keys[e]||, 1e-12)
__global__ void normkeys_kernel(const float* __restrict__ keys,
                                float* __restrict__ kn) {
  __shared__ float red[256];
  int e = blockIdx.x;
  const float* kr = keys + (size_t)e * DIM;
  float s = 0.f;
  for (int i = threadIdx.x; i < DIM; i += 256) { float v = kr[i]; s += v * v; }
  red[threadIdx.x] = s; __syncthreads();
  for (int o = 128; o > 0; o >>= 1) {
    if (threadIdx.x < o) red[threadIdx.x] += red[threadIdx.x + o];
    __syncthreads();
  }
  float scale = 1.f / fmaxf(sqrtf(red[0]), 1e-12f);
  for (int i = threadIdx.x; i < DIM; i += 256)
    kn[(size_t)e * DIM + i] = kr[i] * scale;
}

// ------------------------------------------------------------------
// zero importance / load accumulators in d_out
__global__ void init_kernel(float* imp, int* load) {
  int t = threadIdx.x;
  if (t < NE) { imp[t] = 0.f; load[t] = 0; }
}

// ------------------------------------------------------------------
// bf16 WMMA GEMM:  C[M,N] = act(A[M,K] @ B[N,K]^T + bias)
// block tile 128x128, K-step 32, 8 waves (2x4), wave tile 64x32.
// Tiles staged by CDNA5 async DMA (global_load_async_to_lds_b128),
// double-buffered in LDS: DMA of tile t+1 overlaps WMMA of tile t.
template <bool RELU, bool OUTBF16>
__global__ __launch_bounds__(256)
void gemm_bf16_kernel(const __bf16* __restrict__ A,
                      const __bf16* __restrict__ B,
                      const float* __restrict__ bias,
                      float* __restrict__ outF,
                      __bf16* __restrict__ outH,
                      int M, int N, int K) {
  constexpr int LDT = 40;   // padded bf16 stride (80B rows: 16B-aligned, conflict-free)
  __shared__ __align__(16) __bf16 aT[2][128 * LDT];
  __shared__ __align__(16) __bf16 bT[2][128 * LDT];

  const int tid = threadIdx.x;
  const int lane = tid & 31, w = tid >> 5;
  const int wm = w >> 2, wn = w & 3;      // 2 x 4 wave grid
  const int g = lane >> 4, lh = lane & 15;
  const int m0 = blockIdx.y * 128;
  const int n0 = blockIdx.x * 128;

  v8f acc[4][2] = {};

  // issue one 128x32 bf16 tile pair: 512 16B chunks/matrix, 4 async loads/thread
  auto issue_tile = [&](int buf, int kt) {
    for (int c = tid; c < 512; c += 256) {
      int row = c >> 2, col8 = (c & 3) * 8;
      unsigned la = (unsigned)(uintptr_t)&aT[buf][row * LDT + col8];
      unsigned lb = (unsigned)(uintptr_t)&bT[buf][row * LDT + col8];
      async_load_b128(la, A + (size_t)(m0 + row) * K + kt + col8);
      async_load_b128(lb, B + (size_t)(n0 + row) * K + kt + col8);
    }
  };

  const int nk = K / 32;
  issue_tile(0, 0);
  for (int t = 0; t < nk; ++t) {
    const int buf = t & 1;
    if (t + 1 < nk) {
      issue_tile(buf ^ 1, (t + 1) * 32);
      WAIT_ASYNC_LE4();   // async loads complete in order: current tile landed
    } else {
      WAIT_ASYNC_0();
    }
    __syncthreads();

    // A fragments: ISA 16-bit A 16x32 layout: kbase = 16*(v>>2) + 8*g + 2*(v&3)
    union { v16bf v; unsigned u[8]; } af[4];
    #pragma unroll
    for (int mf = 0; mf < 4; ++mf) {
      int ml = wm * 64 + mf * 16 + lh;
      #pragma unroll
      for (int vv = 0; vv < 8; ++vv) {
        int kb = ((vv >> 2) << 4) + (g << 3) + ((vv & 3) << 1);
        af[mf].u[vv] = *(const unsigned*)&aT[buf][ml * LDT + kb];
      }
    }
    // B fragments: 16-bit B 32x16 layout: kbase = 16*(v>>2) + 8*g + 2*(v&3)
    union { v16bf v; unsigned u[8]; } bfr[2];
    #pragma unroll
    for (int nf = 0; nf < 2; ++nf) {
      int nl = wn * 32 + nf * 16 + lh;
      #pragma unroll
      for (int vv = 0; vv < 8; ++vv) {
        int kb = ((vv >> 2) << 4) + (g << 3) + ((vv & 3) << 1);
        bfr[nf].u[vv] = *(const unsigned*)&bT[buf][nl * LDT + kb];
      }
    }
    #pragma unroll
    for (int mf = 0; mf < 4; ++mf)
      #pragma unroll
      for (int nf = 0; nf < 2; ++nf)
        acc[mf][nf] = __builtin_amdgcn_wmma_f32_16x16x32_bf16(
            false, af[mf].v, false, bfr[nf].v, (short)0, acc[mf][nf],
            false, false);
    __syncthreads();
  }

  // epilogue: C layout row = r + 8*g, col = lane&15
  #pragma unroll
  for (int mf = 0; mf < 4; ++mf)
    #pragma unroll
    for (int nf = 0; nf < 2; ++nf)
      #pragma unroll
      for (int r = 0; r < 8; ++r) {
        int row = m0 + wm * 64 + mf * 16 + g * 8 + r;
        int col = n0 + wn * 32 + nf * 16 + lh;
        float v = acc[mf][nf][r] + bias[col];
        if (RELU) v = fmaxf(v, 0.f);
        if (OUTBF16) outH[(size_t)row * N + col] = (__bf16)v;
        else         outF[(size_t)row * N + col] = v;
      }
}

// ------------------------------------------------------------------
// per-row sum of squares of p
__global__ void rownorm_kernel(const float* __restrict__ p,
                               float* __restrict__ rn) {
  __shared__ float red[256];
  int row = blockIdx.x;
  const float* pr = p + (size_t)row * DIM;
  float s = 0.f;
  for (int i = threadIdx.x; i < DIM; i += 256) { float v = pr[i]; s += v * v; }
  red[threadIdx.x] = s; __syncthreads();
  for (int o = 128; o > 0; o >>= 1) {
    if (threadIdx.x < o) red[threadIdx.x] += red[threadIdx.x + o];
    __syncthreads();
  }
  if (threadIdx.x == 0) rn[row] = red[0];
}

// ------------------------------------------------------------------
// logits[T,64] = (p @ kn^T) / max(||p_row||,1e-12)  — exact f32 WMMA
// 8 waves/block, each wave does 16 rows x 64 experts; kn chunk in LDS.
__global__ __launch_bounds__(256)
void logits_kernel(const float* __restrict__ p,
                   const float* __restrict__ kn,
                   const float* __restrict__ rn,
                   float* __restrict__ logits) {
  constexpr int CH = 128, LDK = 130;      // padded f32 stride (even -> 8B-aligned pairs)
  __shared__ float kT[NE * LDK];
  const int tid = threadIdx.x;
  const int lane = tid & 31, w = tid >> 5;
  const int g = lane >> 4, lh = lane & 15;
  const int m0 = (blockIdx.x * 8 + w) * 16;

  v8f acc[4] = {};

  for (int kc = 0; kc < DIM; kc += CH) {
    for (int c = tid; c < NE * (CH / 2); c += 256) {   // 4096 float2 chunks
      int row = c >> 5, col2 = (c & 31) * 2;
      *(v2f*)&kT[row * LDK + col2] =
          *(const v2f*)&kn[(size_t)row * DIM + kc + col2];
    }
    __syncthreads();
    for (int k0 = 0; k0 < CH; k0 += 4) {
      // f32 A 16x4: lane holds A[m][k0+2g], A[m][k0+2g+1]
      v2f a = *(const v2f*)(p + (size_t)(m0 + lh) * DIM + kc + k0 + 2 * g);
      #pragma unroll
      for (int nt = 0; nt < 4; ++nt) {
        v2f b = *(const v2f*)&kT[(nt * 16 + lh) * LDK + k0 + 2 * g];
        acc[nt] = __builtin_amdgcn_wmma_f32_16x16x4_f32(
            false, a, false, b, (short)0, acc[nt], false, false);
      }
    }
    __syncthreads();
  }

  #pragma unroll
  for (int nt = 0; nt < 4; ++nt)
    #pragma unroll
    for (int r = 0; r < 8; ++r) {
      int row = m0 + g * 8 + r;
      float scale = 1.f / fmaxf(sqrtf(rn[row]), 1e-12f);
      logits[(size_t)row * NE + nt * 16 + lh] = acc[nt][r] * scale;
    }
}

// ------------------------------------------------------------------
// per-row top-8 (stable, ties -> lower index), softmax, stats atomics
__global__ void topk_kernel(const float* __restrict__ logits,
                            int* __restrict__ outIdx,
                            float* __restrict__ outSc,
                            float* __restrict__ imp,
                            int* __restrict__ load) {
  int row = blockIdx.x * blockDim.x + threadIdx.x;
  if (row >= T_TOK) return;
  const float* lg = logits + (size_t)row * NE;
  float tv[NA]; int ti[NA];
  #pragma unroll
  for (int j = 0; j < NA; ++j) { tv[j] = -3.4e38f; ti[j] = -1; }
  for (int e = 0; e < NE; ++e) {
    float v = lg[e];
    if (v > tv[NA - 1]) {
      int j = NA - 1;
      while (j > 0 && v > tv[j - 1]) { tv[j] = tv[j - 1]; ti[j] = ti[j - 1]; --j; }
      tv[j] = v; ti[j] = e;
    }
  }
  float mx = tv[0], s = 0.f, ex[NA];
  #pragma unroll
  for (int j = 0; j < NA; ++j) { ex[j] = expf(tv[j] - mx); s += ex[j]; }
  float inv = 1.f / s;
  #pragma unroll
  for (int j = 0; j < NA; ++j) {
    float sc = ex[j] * inv;
    outIdx[row * NA + j] = ti[j];
    outSc[row * NA + j] = sc;
    atomicAdd(&imp[ti[j]], sc);
    atomicAdd(&load[ti[j]], 1);
  }
}

// ------------------------------------------------------------------
// selected_keys gather: 1.07 GB stream out of L2-resident kn
__global__ void gather_kernel(const int* __restrict__ idx,
                              const float* __restrict__ kn,
                              float* __restrict__ out) {
  int row = blockIdx.x;
  #pragma unroll
  for (int j = 0; j < NA; ++j) {
    int e = idx[row * NA + j];
    const float* src = kn + (size_t)e * DIM;
    float* dst = out + ((size_t)row * NA + j) * DIM;
    for (int t = threadIdx.x; t < DIM; t += 256) dst[t] = src[t];
  }
}

// ------------------------------------------------------------------
// balance loss = 0.01 * (cv2(importance) + cv2(load)), ddof=1
__global__ void balance_kernel(const float* __restrict__ imp,
                               const int* __restrict__ load,
                               float* __restrict__ bl) {
  if (threadIdx.x == 0) {
    float m1 = 0.f, m2 = 0.f;
    for (int i = 0; i < NE; ++i) { m1 += imp[i]; m2 += (float)load[i]; }
    m1 /= NE; m2 /= NE;
    float v1 = 0.f, v2 = 0.f;
    for (int i = 0; i < NE; ++i) {
      float d1 = imp[i] - m1;           v1 += d1 * d1;
      float d2 = (float)load[i] - m2;   v2 += d2 * d2;
    }
    v1 /= (NE - 1); v2 /= (NE - 1);
    *bl = 0.01f * (v1 / (m1 * m1 + 1e-10f) + v2 / (m2 * m2 + 1e-10f));
  }
}

// ------------------------------------------------------------------
extern "C" void kernel_launch(void* const* d_in, const int* in_sizes, int n_in,
                              void* d_out, int out_size, void* d_ws, size_t ws_size,
                              hipStream_t stream) {
  const float* x    = (const float*)d_in[0];
  const float* W1   = (const float*)d_in[1];
  const float* b1   = (const float*)d_in[2];
  const float* W2   = (const float*)d_in[3];
  const float* b2   = (const float*)d_in[4];
  const float* keys = (const float*)d_in[5];

  char* ws = (char*)d_ws;
  __bf16* xb  = (__bf16*)(ws + WS_XB);
  __bf16* W1b = (__bf16*)(ws + WS_W1B);
  __bf16* W2b = (__bf16*)(ws + WS_W2B);
  __bf16* hb  = (__bf16*)(ws + WS_HB);
  float*  p   = (float*)(ws + WS_P);
  float*  kn  = (float*)(ws + WS_KN);
  float*  lg  = (float*)(ws + WS_LG);
  float*  rn  = (float*)(ws + WS_RN);

  float* out  = (float*)d_out;
  int*   oIdx = (int*)(out + O_IDX);
  float* oSc  = out + O_SC;
  float* oBl  = out + O_BL;
  int*   oLd  = (int*)(out + O_LD);
  float* oImp = out + O_IMP;
  float* oSk  = out + O_SK;

  // precision converts
  cvt_bf16_kernel<<<(T_TOK * DIM) / 256, 256, 0, stream>>>(x, xb, T_TOK * DIM);
  cvt_bf16_kernel<<<(DIM * DIM) / 256, 256, 0, stream>>>(W1, W1b, DIM * DIM);
  cvt_bf16_kernel<<<(DIM * DIM) / 256, 256, 0, stream>>>(W2, W2b, DIM * DIM);
  normkeys_kernel<<<NE, 256, 0, stream>>>(keys, kn);
  init_kernel<<<1, 64, 0, stream>>>(oImp, oLd);

  // h = relu(x @ W1^T + b1)  (bf16 out)
  gemm_bf16_kernel<true, true><<<dim3(DIM / 128, T_TOK / 128), 256, 0, stream>>>(
      xb, W1b, b1, nullptr, hb, T_TOK, DIM, DIM);
  // p = h @ W2^T + b2  (f32 out)
  gemm_bf16_kernel<false, false><<<dim3(DIM / 128, T_TOK / 128), 256, 0, stream>>>(
      hb, W2b, b2, p, nullptr, T_TOK, DIM, DIM);

  rownorm_kernel<<<T_TOK, 256, 0, stream>>>(p, rn);
  logits_kernel<<<T_TOK / 128, 256, 0, stream>>>(p, kn, rn, lg);
  topk_kernel<<<T_TOK / 256, 256, 0, stream>>>(lg, oIdx, oSc, oImp, oLd);
  gather_kernel<<<T_TOK, 256, 0, stream>>>(oIdx, kn, oSk);
  balance_kernel<<<1, 64, 0, stream>>>(oImp, oLd, oBl);
}
